// CoverageAttention_88175678587055
// MI455X (gfx1250) — matrix operand order repaired and checked by reading, not compiled, not run on hardware
//
#include <hip/hip_runtime.h>

#define B_ 64
#define L_ 2048
#define F_ 512   // FEAT (= K of big GEMM)
#define U_ 512   // UNITS (= N of big GEMM)
#define H_ 256   // HID

#define BM 128        // rows (l) per block
#define BN 64         // cols (u) per block
#define KC 64         // K chunk staged in LDS
#define SASTR 72      // ushort stride (mult of 8 -> 16B aligned rows; 144B spreads banks)
#define SBSTR 72
#define NBLK (U_ / BN)   // 8
#define CSPLIT 16        // L splits for context partials

#define GLOBAL_AS __attribute__((address_space(1)))
#define LDS_AS    __attribute__((address_space(3)))

typedef __attribute__((ext_vector_type(16))) __bf16 v16bf;
typedef __attribute__((ext_vector_type(2)))  __bf16 v2bf;
typedef __attribute__((ext_vector_type(8)))  float  v8f;
typedef __attribute__((vector_size(16)))     int    vi4;

typedef GLOBAL_AS vi4* gv4p;   // global int4*
typedef LDS_AS    vi4* lv4p;   // LDS int4*

union Frag { uint4 q[2]; v16bf v; };
static_assert(sizeof(Frag) == 32, "frag size");

#if __has_builtin(__builtin_amdgcn_global_load_async_to_lds_b128)
#define ASYNC_B 1
#else
#define ASYNC_B 0
#endif

__device__ __forceinline__ void wait_async0() {
#if __has_builtin(__builtin_amdgcn_s_wait_asynccnt)
  __builtin_amdgcn_s_wait_asynccnt(0);
#else
  asm volatile("s_wait_asynccnt 0x0" ::: "memory");
#endif
}

__device__ __forceinline__ float fast_tanh(float x) {
#if __has_builtin(__builtin_amdgcn_tanhf)
  return __builtin_amdgcn_tanhf(x);
#elif __has_builtin(__builtin_amdgcn_tanh_f32)
  return __builtin_amdgcn_tanh_f32(x);
#else
  return tanhf(x);
#endif
}

// pack two f32 into one dword of bf16 (RNE), preferring the hw packed cvt
__device__ __forceinline__ unsigned pack_bf16(float x, float y) {
#if __has_builtin(__builtin_amdgcn_cvt_pk_bf16_f32)
  auto r = __builtin_amdgcn_cvt_pk_bf16_f32(x, y);
  unsigned u;
  __builtin_memcpy(&u, &r, 4);
  return u;
#else
  union { v2bf h; unsigned u; } cv;
  cv.h = (v2bf){(__bf16)x, (__bf16)y};
  return cv.u;
#endif
}

// ---------- kernel 1: W1 (K x U, f32) -> bf16 transposed [u][k] ----------
__global__ void w1t_kernel(const float* __restrict__ W1, unsigned short* __restrict__ W1t) {
  int u = blockIdx.x, k = threadIdx.x * 2;
  unsigned pk = pack_bf16(W1[(size_t)k * U_ + u], W1[(size_t)(k + 1) * U_ + u]);
  *(unsigned*)&W1t[(size_t)u * F_ + k] = pk;
}

// ---------- kernel 2: hb[b][u] = hidden@W2 + W2_b + W1_b + Uf_b ----------
__global__ void hb_kernel(const float* __restrict__ hidden, const float* __restrict__ W2,
                          const float* __restrict__ W2b, const float* __restrict__ W1b,
                          const float* __restrict__ Ufb, float* __restrict__ hb) {
  int b = blockIdx.y;
  int u = blockIdx.x * 256 + threadIdx.x;
  float acc = W2b[u] + W1b[u] + Ufb[u];
  const float* hrow = hidden + (size_t)b * H_;
  for (int h = 0; h < H_; ++h) acc = fmaf(hrow[h], W2[(size_t)h * U_ + u], acc);
  hb[(size_t)b * U_ + u] = acc;
}

// ---------- kernel 3: bf16 WMMA GEMM + fused tanh/score partials ----------
__global__ __launch_bounds__(256) void score_kernel(
    const float* __restrict__ features, const float* __restrict__ past_att,
    const unsigned short* __restrict__ W1t, const float* __restrict__ hb,
    const float* __restrict__ Ufw, const float* __restrict__ Vw,
    float* __restrict__ score_part) {
  __shared__ __align__(16) unsigned short sA[BM * SASTR];
  __shared__ __align__(16) unsigned short sB[BN * SBSTR];
  __shared__ float spa[BM];
  __shared__ float ssc[BM];

  const int t = threadIdx.x;
  const int ltile = blockIdx.x, nblk = blockIdx.y, b = blockIdx.z;
  const int lbase = ltile * BM;
  const int ubase = nblk * BN;

  if (t < BM) { spa[t] = past_att[(size_t)b * L_ + lbase + t]; ssc[t] = 0.f; }

  const int lane = t & 31, w = t >> 5;
  const int mq = w & 3;        // row quarter (32 rows)
  const int nh = w >> 2;       // col half (32 cols)
  const int ln = lane & 15, hi = lane >> 4;

  v8f acc[2][2] = {};

  const size_t featBase = ((size_t)b * L_ + lbase) * F_;

  for (int kc = 0; kc < F_ / KC; ++kc) {
    __syncthreads();
    // stage A: 128 x 64 f32 -> bf16.  Batch all 8 b128 loads first (8 in
    // flight per thread), then packed-cvt + ds_store_b64.
    {
      float4 va[8];
      for (int i = 0; i < 8; ++i) {
        int idx4 = t + i * 256;               // 0..2047 (float4 index)
        int r = idx4 >> 4, kq = idx4 & 15;
        va[i] = *(const float4*)(features + featBase + (size_t)r * F_ + kc * KC + kq * 4);
      }
      for (int i = 0; i < 8; ++i) {
        int idx4 = t + i * 256;
        int r = idx4 >> 4, kq = idx4 & 15;
        uint2 pk;
        pk.x = pack_bf16(va[i].x, va[i].y);
        pk.y = pack_bf16(va[i].z, va[i].w);
        *(uint2*)&sA[r * SASTR + kq * 4] = pk;
      }
    }
    // stage B: 64 x 64 bf16 from W1t ([u][k], contiguous 16B runs) via async-to-LDS
#if ASYNC_B
    for (int i = 0; i < 2; ++i) {
      int id = t + i * 256;                   // 0..511 (512 x 16B transfers)
      int n = id >> 2, kq = id & 3;
      const unsigned short* src = W1t + ((size_t)(ubase + n)) * F_ + kc * KC + kq * 8;
      // AS1 ptr via integer cast (global flat == AS1); AS3 ptr via low-32-bit
      // truncation (ISA 10.2: LDS aperture uses addr[31:0] directly).
      gv4p gsrc = (gv4p)(unsigned long long)(const void*)src;
      lv4p ldst = (lv4p)(unsigned)(unsigned long long)(void*)&sB[n * SBSTR + kq * 8];
      __builtin_amdgcn_global_load_async_to_lds_b128(gsrc, ldst, 0, 0);
    }
#else
    {
      const uint2* W1tu2 = (const uint2*)W1t;   // 4 ushorts per elem
      uint2 vb[4];
      for (int i = 0; i < 4; ++i) {
        int idx = t + i * 256;                // 0..1023 (uint2 index)
        int n = idx >> 4, kq2 = idx & 15;     // 16 x uint2 per 64-k row... (64*2B/8B=16)
        vb[i] = W1tu2[(((size_t)(ubase + n)) * F_ + kc * KC) / 4 + kq2];
      }
      for (int i = 0; i < 4; ++i) {
        int idx = t + i * 256;
        int n = idx >> 4, kq2 = idx & 15;
        *(uint2*)&sB[n * SBSTR + kq2 * 4] = vb[i];
      }
    }
#endif
    if (kc + 1 < F_ / KC && t < BM)   // global_prefetch_b8 of next K chunk
      __builtin_prefetch(features + featBase + (size_t)t * F_ + (kc + 1) * KC, 0, 1);
#if ASYNC_B
    wait_async0();
#endif
    __syncthreads();

    for (int ki = 0; ki < 2; ++ki) {
      const int k0 = ki * 32;
      Frag a[2], bm[2];
      for (int mi = 0; mi < 2; ++mi) {       // A layout: e<8 -> K=8*hi+e ; e>=8 -> K=16+8*hi+(e-8)
        int m = mq * 32 + mi * 16 + ln;
        a[mi].q[0] = *(const uint4*)&sA[m * SASTR + k0 + 8 * hi];
        a[mi].q[1] = *(const uint4*)&sA[m * SASTR + k0 + 16 + 8 * hi];
      }
      for (int ni = 0; ni < 2; ++ni) {       // B layout: e<8 -> K=8*hi+e ; e>=8 -> K=16+8*hi+(e-8)
        int n = nh * 32 + ni * 16 + ln;
        bm[ni].q[0] = *(const uint4*)&sB[n * SBSTR + k0 + 8 * hi];
        bm[ni].q[1] = *(const uint4*)&sB[n * SBSTR + k0 + 16 + 8 * hi];
      }
      for (int mi = 0; mi < 2; ++mi)
        for (int ni = 0; ni < 2; ++ni)
          acc[mi][ni] = __builtin_amdgcn_wmma_f32_16x16x32_bf16(
              false, a[mi].v, false, bm[ni].v, (short)0, acc[mi][ni], false, false);
    }
  }

  // fused epilogue: + hb + pa*Uf, tanh, * V_w, reduce over this block's 64 cols
  float hbv[2], ufv[2], vwv[2];
  for (int ni = 0; ni < 2; ++ni) {
    int u = ubase + nh * 32 + ni * 16 + ln;
    hbv[ni] = hb[(size_t)b * U_ + u];
    ufv[ni] = Ufw[u];
    vwv[ni] = Vw[u];
  }
  for (int mi = 0; mi < 2; ++mi) {
    float s[8];
    for (int r = 0; r < 8; ++r) {
      int m = mq * 32 + mi * 16 + r + 8 * hi;   // C/D layout: VGPR r, lane-half hi
      float pav = spa[m];
      float c = 0.f;
      for (int ni = 0; ni < 2; ++ni) {
        float v = acc[mi][ni][r] + hbv[ni] + pav * ufv[ni];
        c += fast_tanh(v) * vwv[ni];
      }
      s[r] = c;
    }
    for (int off = 1; off < 16; off <<= 1)
      for (int r = 0; r < 8; ++r) s[r] += __shfl_xor(s[r], off, 16);
    if (ln == 0)
      for (int r = 0; r < 8; ++r)
        atomicAdd(&ssc[mq * 32 + mi * 16 + r + 8 * hi], s[r]);   // ds_add_f32
  }
  __syncthreads();
  if (t < BM)
    score_part[((size_t)b * NBLK + nblk) * L_ + lbase + t] = ssc[t];
}

// ---------- kernel 4: reduce N-block partials + stable softmax over L ----------
__global__ void softmax_kernel(const float* __restrict__ score_part, float* __restrict__ out_w) {
  __shared__ float red[256];
  const int b = blockIdx.x, t = threadIdx.x;
  float v[8];
  for (int i = 0; i < 8; ++i) {
    int l = t + i * 256;
    float s = 0.f;
    for (int nb = 0; nb < NBLK; ++nb) s += score_part[((size_t)b * NBLK + nb) * L_ + l];
    v[i] = s;
  }
  float mx = v[0];
  for (int i = 1; i < 8; ++i) mx = fmaxf(mx, v[i]);
  red[t] = mx; __syncthreads();
  for (int off = 128; off > 0; off >>= 1) {
    if (t < off) red[t] = fmaxf(red[t], red[t + off]);
    __syncthreads();
  }
  mx = red[0]; __syncthreads();
  float sum = 0.f;
  for (int i = 0; i < 8; ++i) { v[i] = __expf(v[i] - mx); sum += v[i]; }
  red[t] = sum; __syncthreads();
  for (int off = 128; off > 0; off >>= 1) {
    if (t < off) red[t] += red[t + off];
    __syncthreads();
  }
  float inv = 1.f / red[0];
  for (int i = 0; i < 8; ++i) out_w[(size_t)b * L_ + t + i * 256] = v[i] * inv;
}

// ---------- kernel 5: context partials over L splits ----------
__global__ void ctx_part_kernel(const float* __restrict__ features, const float* __restrict__ wgt,
                                float* __restrict__ ctx_part) {
  __shared__ float sw[L_ / CSPLIT];
  const int split = blockIdx.x, b = blockIdx.y, t = threadIdx.x;
  const int lc = L_ / CSPLIT;               // 128
  const int l0 = split * lc;
  if (t < lc) sw[t] = wgt[(size_t)b * L_ + l0 + t];
  __syncthreads();
  float acc = 0.f;
  const float* fb = features + ((size_t)b * L_ + l0) * F_;
  for (int l = 0; l < lc; ++l) acc = fmaf(sw[l], fb[(size_t)l * F_ + t], acc);
  ctx_part[((size_t)b * CSPLIT + split) * F_ + t] = acc;
}

// ---------- kernel 6: final context reduction ----------
__global__ void ctx_reduce_kernel(const float* __restrict__ ctx_part, float* __restrict__ out_ctx) {
  const int b = blockIdx.x, t = threadIdx.x;
  float acc = 0.f;
  for (int s = 0; s < CSPLIT; ++s) acc += ctx_part[((size_t)b * CSPLIT + s) * F_ + t];
  out_ctx[(size_t)b * F_ + t] = acc;
}

extern "C" void kernel_launch(void* const* d_in, const int* in_sizes, int n_in,
                              void* d_out, int out_size, void* d_ws, size_t ws_size,
                              hipStream_t stream) {
  (void)in_sizes; (void)n_in; (void)out_size; (void)ws_size;
  const float* features = (const float*)d_in[0];
  const float* hidden   = (const float*)d_in[1];
  const float* past_att = (const float*)d_in[2];
  const float* W1w = (const float*)d_in[3];
  const float* W1b = (const float*)d_in[4];
  const float* W2w = (const float*)d_in[5];
  const float* W2b = (const float*)d_in[6];
  const float* Ufw = (const float*)d_in[7];
  const float* Ufb = (const float*)d_in[8];
  const float* Vw  = (const float*)d_in[9];
  // d_in[10] = V_b: uniform additive shift on scores -> softmax-invariant; omitted.

  // workspace layout (bytes): W1t bf16 512K | hb 128K | score_part 4M | ctx_part 2M
  char* ws = (char*)d_ws;
  unsigned short* W1t  = (unsigned short*)(ws);
  float* hb            = (float*)(ws + 512ull * 1024);
  float* score_part    = (float*)(ws + 640ull * 1024);
  float* ctx_part      = (float*)(ws + 640ull * 1024 + (size_t)B_ * NBLK * L_ * 4);

  float* out_ctx = (float*)d_out;                       // (B, F)
  float* out_w   = (float*)d_out + (size_t)B_ * F_;     // (B, L, 1)

  w1t_kernel<<<dim3(U_), dim3(F_ / 2), 0, stream>>>(W1w, W1t);
  hb_kernel<<<dim3(U_ / 256, B_), dim3(256), 0, stream>>>(hidden, W2w, W2b, W1b, Ufb, hb);
  score_kernel<<<dim3(L_ / BM, NBLK, B_), dim3(256), 0, stream>>>(
      features, past_att, W1t, hb, Ufw, Vw, score_part);
  softmax_kernel<<<dim3(B_), dim3(256), 0, stream>>>(score_part, out_w);
  ctx_part_kernel<<<dim3(CSPLIT, B_), dim3(F_), 0, stream>>>(features, out_w, ctx_part);
  ctx_reduce_kernel<<<dim3(B_), dim3(F_), 0, stream>>>(ctx_part, out_ctx);
}
